// SequentialGPTOSSMoE_75952201663010
// MI455X (gfx1250) — matrix operand order repaired and checked by reading, not compiled, not run on hardware
//
#include <hip/hip_runtime.h>
#include <hip/hip_bf16.h>

// Problem constants (GPT-OSS MoE)
#define HDIM   2048
#define IDIM   2048
#define NEXP   16
#define TOPK   4
#define NTOK   2048
#define ALPHA  1.702f
#define LIMIT  7.0f

// GEMM tiling
#define BM  128   // token rows per block
#define BN  64    // output cols per block
#define BK  32    // K step (matches wmma 16x16x32 bf16)
#define PIT 40    // LDS row pitch in bf16 units (80B, 16B-aligned, conflict-friendly)

typedef __attribute__((ext_vector_type(16))) __bf16 bf16x16;
typedef __attribute__((ext_vector_type(8)))  float  floatx8;

union Frag { bf16x16 v; uint4 u[2]; };
union Acc  { floatx8 v; float f[8]; };

__device__ __forceinline__ unsigned short f2bf(float x) {
  unsigned int u = __float_as_uint(x);
  unsigned int r = u + 0x7FFFu + ((u >> 16) & 1u);   // round-to-nearest-even
  return (unsigned short)(r >> 16);
}

// CDNA5 async copy: global -> LDS, 16 bytes per lane, tracked by ASYNCcnt.
// lds dest = low 32 bits of generic pointer to __shared__ (LDS offset).
__device__ __forceinline__ void async_copy_b128_to_lds(void* lds_ptr, const void* gptr) {
  unsigned lds_off = (unsigned)(unsigned long long)lds_ptr;
  unsigned long long ga = (unsigned long long)gptr;
  asm volatile("global_load_async_to_lds_b128 %0, %1, off"
               :: "v"(lds_off), "v"(ga) : "memory");
}
__device__ __forceinline__ void wait_asynccnt0() {
  asm volatile("s_wait_asynccnt 0x0" ::: "memory");
}

// A fragment (16x32 bf16): lane L holds row m0+(L%16); ks = half*8+{0..7} and 16+half*8+{0..7}
__device__ __forceinline__ bf16x16 load_a_frag(const unsigned short* ldsA, int m0) {
  int lane = threadIdx.x & 31;
  const unsigned short* row = ldsA + (m0 + (lane & 15)) * PIT;
  int half = (lane >> 4) * 8;
  Frag f;
  f.u[0] = *(const uint4*)(row + half);
  f.u[1] = *(const uint4*)(row + 16 + half);
  return f.v;
}

// B fragment (32x16 bf16, LDS stored transposed [n][k]): lane L holds col n0+(L%16); ks = half*8+{0..7} and 16+half*8+{0..7}
__device__ __forceinline__ bf16x16 load_b_frag(const unsigned short* ldsB, int n0) {
  int lane = threadIdx.x & 31;
  const unsigned short* row = ldsB + (n0 + (lane & 15)) * PIT;
  int half = (lane >> 4) * 8;
  Frag f;
  f.u[0] = *(const uint4*)(row + half);
  f.u[1] = *(const uint4*)(row + 16 + half);
  return f.v;
}

// ---------------- Kernel 1: router (logits -> top4 -> softmax -> scatter) ----------------
__global__ __launch_bounds__(256) void moe_router_kernel(
    const float* __restrict__ x, const float* __restrict__ rw, const float* __restrict__ rb,
    float* __restrict__ scores_out, float* __restrict__ probs, int* __restrict__ tidx) {
  int n = blockIdx.x;
  int t = threadIdx.x;
  int e = t & 15, j = t >> 4;
  __shared__ float part[16][17];
  __shared__ float logits[16];
  const float* xr = x + (size_t)n * HDIM;
  const float* wr = rw + (size_t)e * HDIM;
  float s = 0.f;
  for (int i = j; i < HDIM; i += 16) s += xr[i] * wr[i];
  part[e][j] = s;
  __syncthreads();
  if (t < 16) {
    float l = rb[t];
    #pragma unroll
    for (int jj = 0; jj < 16; ++jj) l += part[t][jj];
    logits[t] = l;
  }
  __syncthreads();
  if (t == 0) {
    float v[16];
    #pragma unroll
    for (int i = 0; i < 16; ++i) v[i] = logits[i];
    int   idx[TOPK];
    float val[TOPK];
    #pragma unroll
    for (int k = 0; k < TOPK; ++k) {
      int bi = 0; float bv = -3.4e38f;
      #pragma unroll
      for (int i = 0; i < 16; ++i) { if (v[i] > bv) { bv = v[i]; bi = i; } }
      idx[k] = bi; val[k] = bv; v[bi] = -3.4e38f;
    }
    float mx = val[0];
    float sum = 0.f, p[TOPK];
    #pragma unroll
    for (int k = 0; k < TOPK; ++k) { p[k] = __expf(val[k] - mx); sum += p[k]; }
    float srow[16];
    #pragma unroll
    for (int i = 0; i < 16; ++i) srow[i] = 0.f;
    #pragma unroll
    for (int k = 0; k < TOPK; ++k) {
      p[k] /= sum;
      srow[idx[k]] = p[k];
      probs[n * TOPK + k] = p[k];
      tidx[n * TOPK + k]  = idx[k];
    }
    #pragma unroll
    for (int i = 0; i < 16; ++i) scores_out[(size_t)n * NEXP + i] = srow[i];
  }
}

// ---------------- Kernel 2: deterministic per-expert token lists ----------------
__global__ __launch_bounds__(256) void moe_lists_kernel(
    const int* __restrict__ tidx, int* __restrict__ counts, int* __restrict__ lists) {
  int e = blockIdx.x;
  int t = threadIdx.x;
  const int TPT = NTOK / 256;            // tokens per thread
  __shared__ int pre[256];
  int cnt = 0;
  for (int n = t * TPT; n < t * TPT + TPT; ++n)
    #pragma unroll
    for (int k = 0; k < TOPK; ++k)
      if (tidx[n * TOPK + k] == e) ++cnt;
  pre[t] = cnt;
  __syncthreads();
  for (int off = 1; off < 256; off <<= 1) {
    int v = (t >= off) ? pre[t - off] : 0;
    __syncthreads();
    pre[t] += v;
    __syncthreads();
  }
  int base = pre[t] - cnt;
  int* le = lists + e * NTOK;
  for (int n = t * TPT; n < t * TPT + TPT; ++n)
    #pragma unroll
    for (int k = 0; k < TOPK; ++k)
      if (tidx[n * TOPK + k] == e) le[base++] = n * TOPK + k;   // slot row = token*4+k
  if (t == 255) counts[e] = pre[255];
}

// ---------------- Kernel 3: fused gate/up GEMM + GLU activation (bf16 WMMA) ----------------
__global__ __launch_bounds__(256) void moe_gateup_kernel(
    const float* __restrict__ x,
    const float* __restrict__ gate_w, const float* __restrict__ gate_b,
    const float* __restrict__ up_w,   const float* __restrict__ up_b,
    const float* __restrict__ probs,
    const int* __restrict__ counts, const int* __restrict__ lists,
    unsigned short* __restrict__ act_buf) {
  int e  = blockIdx.z;
  int tb = blockIdx.y * BM;
  int ib = blockIdx.x * BN;
  int count = counts[e];
  if (tb >= count) return;
  int tid = threadIdx.x;

  __shared__ alignas(16) unsigned short sA [BM * PIT];
  __shared__ alignas(16) unsigned short sBg[BN * PIT];
  __shared__ alignas(16) unsigned short sBu[BN * PIT];
  __shared__ alignas(16) unsigned short sAct[BM * BN];
  __shared__ int   s_slot[BM];
  __shared__ float s_prob[BM];
  __shared__ float s_gb[BN];
  __shared__ float s_ub[BN];

  for (int m = tid; m < BM; m += 256) {
    int idx  = tb + m;
    int slot = (idx < count) ? lists[e * NTOK + idx] : -1;
    s_slot[m] = slot;
    s_prob[m] = (slot >= 0) ? probs[slot] : 0.f;
  }
  for (int i = tid; i < BN; i += 256) {
    s_gb[i] = gate_b[(size_t)e * IDIM + ib + i];
    s_ub[i] = up_b  [(size_t)e * IDIM + ib + i];
  }

  int wave = tid >> 5;
  int wm = (wave & 3) * 32;
  int wn = (wave >> 2) * 32;

  Acc gacc[2][2], uacc[2][2];
  #pragma unroll
  for (int a = 0; a < 2; ++a)
    #pragma unroll
    for (int b = 0; b < 2; ++b)
      #pragma unroll
      for (int r = 0; r < 8; ++r) { gacc[a][b].f[r] = 0.f; uacc[a][b].f[r] = 0.f; }

  const float* gw = gate_w + (size_t)e * HDIM * IDIM;
  const float* uw = up_w   + (size_t)e * HDIM * IDIM;

  for (int kb = 0; kb < HDIM; kb += BK) {
    __syncthreads();
    // stage A: 128 token rows x 32 k (gathered, f32 -> bf16)
    for (int q = tid; q < BM * (BK / 4); q += 256) {
      int m  = q >> 3;
      int cg = (q & 7) * 4;
      int slot = s_slot[m];
      float4 v = make_float4(0.f, 0.f, 0.f, 0.f);
      if (slot >= 0) {
        const float* src = x + (size_t)(slot >> 2) * HDIM + kb + cg;
        v = *(const float4*)src;
        if (kb + BK < HDIM) __builtin_prefetch(src + BK, 0, 0);
      }
      unsigned short* dst = sA + m * PIT + cg;
      dst[0] = f2bf(v.x); dst[1] = f2bf(v.y); dst[2] = f2bf(v.z); dst[3] = f2bf(v.w);
    }
    // stage Bg/Bu transposed: lds[n][k] = W[kb+k][ib+n]; prefetch next K tile
    for (int q = tid; q < 2 * BK * (BN / 4); q += 256) {
      int mat = q >> 9;
      int r   = q & 511;
      int k   = r >> 4;
      int cg  = (r & 15) * 4;
      const float* src = (mat ? uw : gw) + (size_t)(kb + k) * IDIM + ib + cg;
      float4 v = *(const float4*)src;
      if (kb + BK < HDIM) __builtin_prefetch(src + (size_t)BK * IDIM, 0, 0);
      unsigned short* db = mat ? sBu : sBg;
      db[(cg + 0) * PIT + k] = f2bf(v.x);
      db[(cg + 1) * PIT + k] = f2bf(v.y);
      db[(cg + 2) * PIT + k] = f2bf(v.z);
      db[(cg + 3) * PIT + k] = f2bf(v.w);
    }
    __syncthreads();
    bf16x16 a0  = load_a_frag(sA,  wm);
    bf16x16 a1  = load_a_frag(sA,  wm + 16);
    bf16x16 bg0 = load_b_frag(sBg, wn);
    bf16x16 bg1 = load_b_frag(sBg, wn + 16);
    bf16x16 bu0 = load_b_frag(sBu, wn);
    bf16x16 bu1 = load_b_frag(sBu, wn + 16);
    gacc[0][0].v = __builtin_amdgcn_wmma_f32_16x16x32_bf16(false, a0, false, bg0, (short)0, gacc[0][0].v, false, false);
    gacc[0][1].v = __builtin_amdgcn_wmma_f32_16x16x32_bf16(false, a0, false, bg1, (short)0, gacc[0][1].v, false, false);
    gacc[1][0].v = __builtin_amdgcn_wmma_f32_16x16x32_bf16(false, a1, false, bg0, (short)0, gacc[1][0].v, false, false);
    gacc[1][1].v = __builtin_amdgcn_wmma_f32_16x16x32_bf16(false, a1, false, bg1, (short)0, gacc[1][1].v, false, false);
    uacc[0][0].v = __builtin_amdgcn_wmma_f32_16x16x32_bf16(false, a0, false, bu0, (short)0, uacc[0][0].v, false, false);
    uacc[0][1].v = __builtin_amdgcn_wmma_f32_16x16x32_bf16(false, a0, false, bu1, (short)0, uacc[0][1].v, false, false);
    uacc[1][0].v = __builtin_amdgcn_wmma_f32_16x16x32_bf16(false, a1, false, bu0, (short)0, uacc[1][0].v, false, false);
    uacc[1][1].v = __builtin_amdgcn_wmma_f32_16x16x32_bf16(false, a1, false, bu1, (short)0, uacc[1][1].v, false, false);
  }
  __syncthreads();

  // epilogue: bias + clamp + GLU, fold combine weight, stage to LDS
  int lane = tid & 31;
  int lcol = lane & 15;
  int lhi  = (lane >> 4) * 8;
  #pragma unroll
  for (int tm = 0; tm < 2; ++tm)
    #pragma unroll
    for (int tn = 0; tn < 2; ++tn)
      #pragma unroll
      for (int r = 0; r < 8; ++r) {
        int row = wm + tm * 16 + lhi + r;
        int col = wn + tn * 16 + lcol;
        float g = gacc[tm][tn].f[r] + s_gb[col];
        float u = uacc[tm][tn].f[r] + s_ub[col];
        g = fminf(g, LIMIT);
        u = fminf(fmaxf(u, -LIMIT), LIMIT);
        float glu = g / (1.f + __expf(-ALPHA * g));
        sAct[row * BN + col] = f2bf((u + 1.f) * glu * s_prob[row]);
      }
  __syncthreads();

  // coalesced bf16 writes of valid rows
  for (int q = tid; q < BM * (BN / 8); q += 256) {
    int m    = q >> 3;
    int part = (q & 7) * 8;
    int slot = s_slot[m];
    if (slot >= 0) {
      uint4 v = *(const uint4*)(sAct + m * BN + part);
      *(uint4*)(act_buf + (size_t)slot * IDIM + ib + part) = v;
    }
  }
}

// ---------------- Kernel 4: down GEMM (bf16 WMMA), async-to-LDS A staging ----------------
__global__ __launch_bounds__(256) void moe_down_kernel(
    const unsigned short* __restrict__ act_buf,
    const float* __restrict__ down_w, const float* __restrict__ down_b,
    const float* __restrict__ probs,
    const int* __restrict__ counts, const int* __restrict__ lists,
    float* __restrict__ out_slot) {
  int e  = blockIdx.z;
  int tb = blockIdx.y * BM;
  int hb = blockIdx.x * BN;
  int count = counts[e];
  if (tb >= count) return;
  int tid = threadIdx.x;

  __shared__ alignas(16) unsigned short sA[BM * PIT];
  __shared__ alignas(16) unsigned short sB[BN * PIT];
  __shared__ int   s_slot[BM];
  __shared__ float s_prob[BM];
  __shared__ float s_db[BN];

  for (int m = tid; m < BM; m += 256) {
    int idx  = tb + m;
    int slot = (idx < count) ? lists[e * NTOK + idx] : -1;
    s_slot[m] = slot;
    s_prob[m] = (slot >= 0) ? probs[slot] : 0.f;
  }
  for (int i = tid; i < BN; i += 256) s_db[i] = down_b[(size_t)e * HDIM + hb + i];

  int wave = tid >> 5;
  int wm = (wave & 3) * 32;
  int wn = (wave >> 2) * 32;

  Acc acc[2][2];
  #pragma unroll
  for (int a = 0; a < 2; ++a)
    #pragma unroll
    for (int b = 0; b < 2; ++b)
      #pragma unroll
      for (int r = 0; r < 8; ++r) acc[a][b].f[r] = 0.f;

  const float* dw = down_w + (size_t)e * IDIM * HDIM;

  for (int kb = 0; kb < IDIM; kb += BK) {
    __syncthreads();
    // stage A (already bf16) via CDNA5 async global->LDS copies (ASYNCcnt path).
    // Padding rows are clamped to row 0 (never escape: epilogue guards on slot).
    for (int q = tid; q < BM * 4; q += 256) {
      int m    = q >> 2;
      int part = (q & 3) * 8;
      int slot = s_slot[m];
      int srow = (slot >= 0) ? slot : 0;
      async_copy_b128_to_lds(sA + m * PIT + part,
                             act_buf + (size_t)srow * IDIM + kb + part);
    }
    // stage B transposed: lds[n][k] = dw[kb+k][hb+n]; prefetch next K tile
    for (int q = tid; q < BK * (BN / 4); q += 256) {
      int k  = q >> 4;
      int cg = (q & 15) * 4;
      const float* src = dw + (size_t)(kb + k) * HDIM + hb + cg;
      float4 v = *(const float4*)src;
      if (kb + BK < IDIM) __builtin_prefetch(src + (size_t)BK * HDIM, 0, 0);
      sB[(cg + 0) * PIT + k] = f2bf(v.x);
      sB[(cg + 1) * PIT + k] = f2bf(v.y);
      sB[(cg + 2) * PIT + k] = f2bf(v.z);
      sB[(cg + 3) * PIT + k] = f2bf(v.w);
    }
    wait_asynccnt0();
    __syncthreads();
    bf16x16 a0 = load_a_frag(sA, wm);
    bf16x16 a1 = load_a_frag(sA, wm + 16);
    bf16x16 b0 = load_b_frag(sB, wn);
    bf16x16 b1 = load_b_frag(sB, wn + 16);
    acc[0][0].v = __builtin_amdgcn_wmma_f32_16x16x32_bf16(false, a0, false, b0, (short)0, acc[0][0].v, false, false);
    acc[0][1].v = __builtin_amdgcn_wmma_f32_16x16x32_bf16(false, a0, false, b1, (short)0, acc[0][1].v, false, false);
    acc[1][0].v = __builtin_amdgcn_wmma_f32_16x16x32_bf16(false, a1, false, b0, (short)0, acc[1][0].v, false, false);
    acc[1][1].v = __builtin_amdgcn_wmma_f32_16x16x32_bf16(false, a1, false, b1, (short)0, acc[1][1].v, false, false);
  }

  int lane = tid & 31;
  int lcol = lane & 15;
  int lhi  = (lane >> 4) * 8;
  #pragma unroll
  for (int tm = 0; tm < 2; ++tm)
    #pragma unroll
    for (int tn = 0; tn < 2; ++tn)
      #pragma unroll
      for (int r = 0; r < 8; ++r) {
        int row  = wm + tm * 16 + lhi + r;
        int col  = wn + tn * 16 + lcol;
        int slot = s_slot[row];
        if (slot >= 0)
          out_slot[(size_t)slot * HDIM + hb + col] = acc[tm][tn].f[r] + s_prob[row] * s_db[col];
      }
}

// ---------------- Kernel 5: reduce 4 slots per token -> out ----------------
__global__ __launch_bounds__(256) void moe_reduce_kernel(
    const float* __restrict__ out_slot, float* __restrict__ out) {
  int idx = blockIdx.x * 256 + threadIdx.x;   // over NTOK*HDIM
  int n = idx / HDIM;
  int h = idx - n * HDIM;
  const float* base = out_slot + (size_t)n * TOPK * HDIM + h;
  float s = base[0] + base[HDIM] + base[2 * HDIM] + base[3 * HDIM];
  out[idx] = s;
}

// ---------------- Host launcher ----------------
extern "C" void kernel_launch(void* const* d_in, const int* in_sizes, int n_in,
                              void* d_out, int out_size, void* d_ws, size_t ws_size,
                              hipStream_t stream) {
  const float* x       = (const float*)d_in[0];
  const float* rw      = (const float*)d_in[1];
  const float* rb      = (const float*)d_in[2];
  const float* gate_w  = (const float*)d_in[3];
  const float* gate_b  = (const float*)d_in[4];
  const float* up_w    = (const float*)d_in[5];
  const float* up_b    = (const float*)d_in[6];
  const float* down_w  = (const float*)d_in[7];
  const float* down_b  = (const float*)d_in[8];

  float* out        = (float*)d_out;                               // [NTOK, HDIM]
  float* scores_out = (float*)d_out + (size_t)NTOK * HDIM;         // [NTOK, NEXP]

  char* ws = (char*)d_ws;
  // workspace layout (bytes); total ~96.3 MB
  float*          probs    = (float*)(ws + 0);                     // 32 KB
  int*            tidx     = (int*)  (ws + 32768);                 // 32 KB
  int*            counts   = (int*)  (ws + 65536);                 // 64 B
  int*            lists    = (int*)  (ws + 66560);                 // 128 KB
  unsigned short* act_buf  = (unsigned short*)(ws + 262144);       // 32 MB  (NTOK*TOPK*IDIM bf16)
  float*          out_slot = (float*)(ws + 262144 + (size_t)NTOK * TOPK * IDIM * 2); // 64 MB

  moe_router_kernel<<<NTOK, 256, 0, stream>>>(x, rw, rb, scores_out, probs, tidx);
  moe_lists_kernel<<<NEXP, 256, 0, stream>>>(tidx, counts, lists);
  moe_gateup_kernel<<<dim3(IDIM / BN, NTOK / BM, NEXP), 256, 0, stream>>>(
      x, gate_w, gate_b, up_w, up_b, probs, counts, lists, act_buf);
  moe_down_kernel<<<dim3(HDIM / BN, NTOK / BM, NEXP), 256, 0, stream>>>(
      act_buf, down_w, down_b, probs, counts, lists, out_slot);
  moe_reduce_kernel<<<(NTOK * HDIM) / 256, 256, 0, stream>>>(out_slot, out);
}
